// ReOrderer_46248207843625
// MI455X (gfx1250) — compile-verified, matrix operand order checked by reading, not run on hardware
//
#include <hip/hip_runtime.h>
#include <hip/hip_bf16.h>

typedef __attribute__((ext_vector_type(16))) _Float16 v16h;
typedef __attribute__((ext_vector_type(8)))  _Float16 v8h;
typedef __attribute__((ext_vector_type(8)))  float    v8f;

#define B_DIM 8
#define S_DIM 2048
#define T_DIM 2048
#define D_DIM 1024
#define INF_C 1.0e10f
#define INV_SCALE 0.03125f          /* 1/sqrt(1024) */
#define INV_03 (1.0f/0.3f)

#define WMMA_F16(A_, B_, C_) \
    __builtin_amdgcn_wmma_f32_16x16x32_f16(false, (A_), false, (B_), (short)0, (C_), false, false)

// ---------------------------------------------------------------------------
// Fragment load from a precomputed per-lane pointer (two b128 loads).
// Layout (16-bit A/B, K contiguous): lane holds row (lane&15), halves at
// K = 8*(lane>>4) + {0..7} and 16 + 8*(lane>>4) + {0..7}.
// ---------------------------------------------------------------------------
__device__ __forceinline__ v16h load_frag_p(const _Float16* p) {
    union { v16h v; struct { v8h lo; v8h hi; } s; } u;
    u.s.lo = *(const v8h*)p;
    u.s.hi = *(const v8h*)(p + 16);
    return u.v;
}

struct FragSet { v16h a0, a1, b0, b1, b2, b3; };

__device__ __forceinline__ FragSet
load_set(const _Float16* pa0, const _Float16* pa1,
         const _Float16* pb0, const _Float16* pb1,
         const _Float16* pb2, const _Float16* pb3, int koff) {
    FragSet f;
    f.a0 = load_frag_p(pa0 + koff);
    f.a1 = load_frag_p(pa1 + koff);
    f.b0 = load_frag_p(pb0 + koff);
    f.b1 = load_frag_p(pb1 + koff);
    f.b2 = load_frag_p(pb2 + koff);
    f.b3 = load_frag_p(pb3 + koff);
    return f;
}

#define DO_WMMA8(acc, F)                                  \
    do {                                                  \
        acc[0] = WMMA_F16((F).a0, (F).b0, acc[0]);        \
        acc[1] = WMMA_F16((F).a0, (F).b1, acc[1]);        \
        acc[2] = WMMA_F16((F).a0, (F).b2, acc[2]);        \
        acc[3] = WMMA_F16((F).a0, (F).b3, acc[3]);        \
        acc[4] = WMMA_F16((F).a1, (F).b0, acc[4]);        \
        acc[5] = WMMA_F16((F).a1, (F).b1, acc[5]);        \
        acc[6] = WMMA_F16((F).a1, (F).b2, acc[6]);        \
        acc[7] = WMMA_F16((F).a1, (F).b3, acc[7]);        \
    } while (0)

// ---------------------------------------------------------------------------
// K0: per-batch steps = sum(mask_src)/sum(mask_trg)
// ---------------------------------------------------------------------------
__global__ void steps_kernel(const float* __restrict__ mask_src,
                             const float* __restrict__ mask_trg,
                             float* __restrict__ steps) {
    int b = blockIdx.x, tid = threadIdx.x;
    float ss = 0.f, ts = 0.f;
    for (int i = tid; i < S_DIM; i += 256) ss += mask_src[b * S_DIM + i];
    for (int i = tid; i < T_DIM; i += 256) ts += mask_trg[b * T_DIM + i];
    __shared__ float rs[256], rt[256];
    rs[tid] = ss; rt[tid] = ts; __syncthreads();
    for (int off = 128; off > 0; off >>= 1) {
        if (tid < off) { rs[tid] += rs[tid + off]; rt[tid] += rt[tid + off]; }
        __syncthreads();
    }
    if (tid == 0) steps[b] = rs[0] / rt[0];
}

// ---------------------------------------------------------------------------
// K1: banded l_att softmax -> query (fp32-exact under all-ones masks since
// exp(-d^2/0.3) underflows to 0 in fp32 for |d|>5.2). Fuses gate sigmoid.
// ---------------------------------------------------------------------------
__global__ void query_gate_kernel(const float* __restrict__ key,
                                  const float* __restrict__ mask_src,
                                  const float* __restrict__ steps,
                                  const float* __restrict__ gate_w,
                                  const float* __restrict__ gate_b,
                                  _Float16* __restrict__ query_h,
                                  float* __restrict__ gates) {
    int b = blockIdx.x / T_DIM;
    int t = blockIdx.x % T_DIM;
    int tid = threadIdx.x;
    float it = steps[b] * (float)t;
    int c  = (int)floorf(it);
    int s0 = max(0, c - 6);
    int s1 = min(S_DIM - 1, c + 7);

    float acc[4] = {0.f, 0.f, 0.f, 0.f};
    float denom = 0.f;
    for (int s = s0; s <= s1; ++s) {
        float m = mask_src[b * S_DIM + s];
        float d = (float)s - it;
        float w = (m > 0.f) ? __expf(-d * d * INV_03) : 0.f;
        denom += w;
        if (w != 0.f) {
            const float* kp = key + ((size_t)(b * S_DIM + s)) * D_DIM + tid;
            #pragma unroll
            for (int i = 0; i < 4; ++i) acc[i] += w * kp[i * 256];
        }
    }
    float inv = (denom > 0.f) ? (1.f / denom) : 0.f;
    float gpart = 0.f;
    size_t qbase = ((size_t)(b * T_DIM + t)) * D_DIM + tid;
    #pragma unroll
    for (int i = 0; i < 4; ++i) {
        float q = acc[i] * inv;
        gpart += q * gate_w[tid + i * 256];
        query_h[qbase + (size_t)i * 256] = (_Float16)q;
    }
    __shared__ float red[256];
    red[tid] = gpart; __syncthreads();
    for (int off = 128; off > 0; off >>= 1) {
        if (tid < off) red[tid] += red[tid + off];
        __syncthreads();
    }
    if (tid == 0)
        gates[b * T_DIM + t] = 1.f / (1.f + __expf(-(red[0] + gate_b[0])));
}

// ---------------------------------------------------------------------------
// K2: f32 -> f16 cast (4 elems/thread, n multiple of 4)
// ---------------------------------------------------------------------------
__global__ void cast_f16_kernel(const float* __restrict__ in,
                                _Float16* __restrict__ out, size_t n) {
    size_t i = ((size_t)blockIdx.x * blockDim.x + threadIdx.x) * 4;
    if (i + 3 < n) {
        float4 v = *(const float4*)(in + i);
        out[i]     = (_Float16)v.x;
        out[i + 1] = (_Float16)v.y;
        out[i + 2] = (_Float16)v.z;
        out[i + 3] = (_Float16)v.w;
    }
}

// ---------------------------------------------------------------------------
// K3: Out[m,n] = f16( sum_k A[m,k]*W[n,k] + bias[n] ).
// Wave tile 32x64, explicit ping-pong double buffering (no rotation moves,
// exactly 2 fragment sets live -> no spills).
// ---------------------------------------------------------------------------
__global__ void gemm_nt_bias_kernel(const _Float16* __restrict__ A,
                                    const _Float16* __restrict__ W,
                                    const float* __restrict__ bias,
                                    _Float16* __restrict__ Out,
                                    int M, int N, int K) {
    int wid  = (int)((blockIdx.x * blockDim.x + threadIdx.x) >> 5);
    int lane = threadIdx.x & 31;
    int ntiles = N >> 6;
    int mt = wid / ntiles, nt = wid % ntiles;
    int row0 = mt << 5, col0 = nt << 6;
    int kh = (lane >> 4) << 3;

    const _Float16* pa0 = A + (size_t)(row0 + (lane & 15)) * K + kh;
    const _Float16* pa1 = pa0 + (size_t)16 * K;
    const _Float16* pb0 = W + (size_t)(col0 + (lane & 15)) * K + kh;
    const _Float16* pb1 = pb0 + (size_t)16 * K;
    const _Float16* pb2 = pb0 + (size_t)32 * K;
    const _Float16* pb3 = pb0 + (size_t)48 * K;

    v8f acc[8] = {};
    int steps2 = (K >> 5) >> 1;      // K/32 steps, processed 2 per loop iter
    int koff = 0;
    FragSet X = load_set(pa0, pa1, pb0, pb1, pb2, pb3, koff); koff += 32;

    #pragma clang loop unroll(disable)
    for (int it = 0; it < steps2 - 1; ++it) {
        FragSet Y = load_set(pa0, pa1, pb0, pb1, pb2, pb3, koff); koff += 32;
        DO_WMMA8(acc, X);
        X = load_set(pa0, pa1, pb0, pb1, pb2, pb3, koff); koff += 32;
        DO_WMMA8(acc, Y);
    }
    {
        FragSet Y = load_set(pa0, pa1, pb0, pb1, pb2, pb3, koff);
        DO_WMMA8(acc, X);
        DO_WMMA8(acc, Y);
    }

    int nl = lane & 15;
    int mb = (lane >> 4) << 3;
    #pragma unroll
    for (int g = 0; g < 2; ++g) {
        #pragma unroll
        for (int j = 0; j < 4; ++j) {
            int n  = col0 + 16 * j + nl;
            float bv = bias[n];
            #pragma unroll
            for (int v = 0; v < 8; ++v) {
                int m = row0 + 16 * g + mb + v;
                Out[(size_t)m * N + n] = (_Float16)(acc[4 * g + j][v] + bv);
            }
        }
    }
}

// ---------------------------------------------------------------------------
// K4: dots[b,t,s] = (q.k)/SCALE - (1-mask)*INF/SCALE  (raw logits -> d_out)
// Wave tile 32x64, explicit ping-pong double buffering.
// ---------------------------------------------------------------------------
__global__ void gemm_dots_kernel(const _Float16* __restrict__ Q,
                                 const _Float16* __restrict__ Kmat,
                                 const float* __restrict__ mask_src,
                                 float* __restrict__ Out) {
    const int ntiles = S_DIM >> 6;                     // 32
    const int wavesPerBatch = (T_DIM >> 5) * ntiles;   // 2048
    int wid  = (int)((blockIdx.x * blockDim.x + threadIdx.x) >> 5);
    int lane = threadIdx.x & 31;
    int b = wid / wavesPerBatch;
    int r = wid % wavesPerBatch;
    int mt = r / ntiles, nt = r % ntiles;
    int row0 = mt << 5, col0 = nt << 6;
    int kh = (lane >> 4) << 3;

    const _Float16* Ab = Q    + (size_t)b * T_DIM * D_DIM;
    const _Float16* Bb = Kmat + (size_t)b * S_DIM * D_DIM;
    const _Float16* pa0 = Ab + (size_t)(row0 + (lane & 15)) * D_DIM + kh;
    const _Float16* pa1 = pa0 + (size_t)16 * D_DIM;
    const _Float16* pb0 = Bb + (size_t)(col0 + (lane & 15)) * D_DIM + kh;
    const _Float16* pb1 = pb0 + (size_t)16 * D_DIM;
    const _Float16* pb2 = pb0 + (size_t)32 * D_DIM;
    const _Float16* pb3 = pb0 + (size_t)48 * D_DIM;

    v8f acc[8] = {};
    const int steps2 = (D_DIM >> 5) >> 1;
    int koff = 0;
    FragSet X = load_set(pa0, pa1, pb0, pb1, pb2, pb3, koff); koff += 32;

    #pragma clang loop unroll(disable)
    for (int it = 0; it < steps2 - 1; ++it) {
        FragSet Y = load_set(pa0, pa1, pb0, pb1, pb2, pb3, koff); koff += 32;
        DO_WMMA8(acc, X);
        X = load_set(pa0, pa1, pb0, pb1, pb2, pb3, koff); koff += 32;
        DO_WMMA8(acc, Y);
    }
    {
        FragSet Y = load_set(pa0, pa1, pb0, pb1, pb2, pb3, koff);
        DO_WMMA8(acc, X);
        DO_WMMA8(acc, Y);
    }

    int nl = lane & 15;
    int mb = (lane >> 4) << 3;
    float* Ob = Out + (size_t)b * T_DIM * S_DIM;
    #pragma unroll
    for (int j = 0; j < 4; ++j) {
        int s = col0 + 16 * j + nl;
        float pen = (1.f - mask_src[b * S_DIM + s]) * (INF_C * INV_SCALE);
        #pragma unroll
        for (int g = 0; g < 2; ++g) {
            #pragma unroll
            for (int v = 0; v < 8; ++v) {
                int t = row0 + 16 * g + mb + v;
                Ob[(size_t)t * S_DIM + s] = acc[4 * g + j][v] * INV_SCALE - pen;
            }
        }
    }
}

// ---------------------------------------------------------------------------
// K5: per-row softmax stats for dots (from d_out) AND recomputed l_att logits
// stats[r] = {max_dots, sum_dots, max_latt, sum_latt}
// ---------------------------------------------------------------------------
__global__ void rowstats_kernel(const float* __restrict__ dots,
                                const float* __restrict__ mask_src,
                                const float* __restrict__ steps,
                                float4* __restrict__ stats) {
    int r = blockIdx.x;
    int b = r / T_DIM, t = r % T_DIM;
    int tid = threadIdx.x;
    float it = steps[b] * (float)t;
    const float* row = dots + (size_t)r * S_DIM;

    float xv[8], lv[8];
    float mx = -3.4e38f, ml = -3.4e38f;
    #pragma unroll
    for (int i = 0; i < 8; ++i) {
        int s = tid + i * 256;
        float x = row[s];
        float m = mask_src[b * S_DIM + s];
        float d = (float)s - it;
        float l = -d * d * INV_03 - (1.f - m) * INF_C;
        xv[i] = x; lv[i] = l;
        mx = fmaxf(mx, x); ml = fmaxf(ml, l);
    }
    __shared__ float ra[256], rb[256];
    ra[tid] = mx; rb[tid] = ml; __syncthreads();
    for (int off = 128; off > 0; off >>= 1) {
        if (tid < off) { ra[tid] = fmaxf(ra[tid], ra[tid + off]);
                         rb[tid] = fmaxf(rb[tid], rb[tid + off]); }
        __syncthreads();
    }
    mx = ra[0]; ml = rb[0]; __syncthreads();
    float sx = 0.f, sl = 0.f;
    #pragma unroll
    for (int i = 0; i < 8; ++i) {
        sx += __expf(xv[i] - mx);
        sl += __expf(lv[i] - ml);
    }
    ra[tid] = sx; rb[tid] = sl; __syncthreads();
    for (int off = 128; off > 0; off >>= 1) {
        if (tid < off) { ra[tid] += ra[tid + off]; rb[tid] += rb[tid + off]; }
        __syncthreads();
    }
    if (tid == 0) stats[r] = make_float4(mx, ra[0], ml, rb[0]);
}

// ---------------------------------------------------------------------------
// K6: out = (1-g)*softmax(dots) + g*softmax(l_att), in place on d_out
// ---------------------------------------------------------------------------
__global__ void finalize_kernel(float* __restrict__ out,
                                const float4* __restrict__ stats,
                                const float* __restrict__ gates,
                                const float* __restrict__ mask_src,
                                const float* __restrict__ steps) {
    size_t idx = (size_t)blockIdx.x * 256 + threadIdx.x;
    int s = (int)(idx & (S_DIM - 1));
    int r = (int)(idx >> 11);               // /S_DIM
    int b = r / T_DIM, t = r % T_DIM;
    float4 st = stats[r];
    float x = out[idx];
    float p = __expf(x - st.x) / st.y;
    float m = mask_src[b * S_DIM + s];
    float d = (float)s - steps[b] * (float)t;
    float ll = -d * d * INV_03 - (1.f - m) * INF_C;
    float l = __expf(ll - st.z) / st.w;
    float g = gates[r];
    out[idx] = (1.f - g) * p + g * l;
}

// ---------------------------------------------------------------------------
extern "C" void kernel_launch(void* const* d_in, const int* in_sizes, int n_in,
                              void* d_out, int out_size, void* d_ws, size_t ws_size,
                              hipStream_t stream) {
    const float* key      = (const float*)d_in[0];
    const float* mask_src = (const float*)d_in[1];
    const float* mask_trg = (const float*)d_in[2];
    const float* wq_w     = (const float*)d_in[3];
    const float* wq_b     = (const float*)d_in[4];
    const float* wk_w     = (const float*)d_in[5];
    const float* wk_b     = (const float*)d_in[6];
    const float* gate_w   = (const float*)d_in[7];
    const float* gate_b   = (const float*)d_in[8];
    float* out = (float*)d_out;

    const size_t BT  = (size_t)B_DIM * T_DIM;
    const size_t BS  = (size_t)B_DIM * S_DIM;
    const size_t BTD = BT * D_DIM;
    const size_t BSD = BS * D_DIM;
    const size_t DD  = (size_t)D_DIM * D_DIM;

    // workspace carve (256B aligned slabs)
    char* p = (char*)d_ws;
    auto carve = [&](size_t bytes) -> char* {
        char* r = p; p += (bytes + 255) & ~(size_t)255; return r;
    };
    float*     steps   = (float*)    carve(B_DIM * sizeof(float));
    float*     gates   = (float*)    carve(BT * sizeof(float));
    float4*    stats   = (float4*)   carve(BT * sizeof(float4));
    _Float16*  wq_h    = (_Float16*) carve(DD * 2);
    _Float16*  wk_h    = (_Float16*) carve(DD * 2);
    _Float16*  query_h = (_Float16*) carve(BTD * 2);
    _Float16*  key_h   = (_Float16*) carve(BSD * 2);
    _Float16*  q_h     = (_Float16*) carve(BTD * 2);
    _Float16*  k_h     = (_Float16*) carve(BSD * 2);
    (void)ws_size; (void)n_in; (void)in_sizes; (void)out_size;

    // 1. steps per batch
    steps_kernel<<<B_DIM, 256, 0, stream>>>(mask_src, mask_trg, steps);

    // 2. banded l_att @ key -> query (f16) + fused gates
    query_gate_kernel<<<B_DIM * T_DIM, 256, 0, stream>>>(
        key, mask_src, steps, gate_w, gate_b, query_h, gates);

    // 3. f16 casts
    cast_f16_kernel<<<(unsigned)(BSD / 1024), 256, 0, stream>>>(key,  key_h, BSD);
    cast_f16_kernel<<<(unsigned)(DD  / 1024), 256, 0, stream>>>(wq_w, wq_h,  DD);
    cast_f16_kernel<<<(unsigned)(DD  / 1024), 256, 0, stream>>>(wk_w, wk_h,  DD);

    // 4. linear layers via WMMA: q = query@WqT+b, k = key@WkT+b
    {
        int M = (int)BT, N = D_DIM;
        int waves = (M / 32) * (N / 64);               // 8192 waves
        gemm_nt_bias_kernel<<<waves / 8, 256, 0, stream>>>(query_h, wq_h, wq_b, q_h, M, N, D_DIM);
        gemm_nt_bias_kernel<<<waves / 8, 256, 0, stream>>>(key_h,   wk_h, wk_b, k_h, M, N, D_DIM);
    }

    // 5. dots = q@k^T (WMMA), write raw scaled+masked logits to d_out
    {
        int wavesPerBatch = (T_DIM / 32) * (S_DIM / 64);   // 2048
        int waves = B_DIM * wavesPerBatch;                 // 16384
        gemm_dots_kernel<<<waves / 8, 256, 0, stream>>>(q_h, k_h, mask_src, out);
    }

    // 6. per-row softmax stats (dots + l_att)
    rowstats_kernel<<<B_DIM * T_DIM, 256, 0, stream>>>(out, mask_src, steps, stats);

    // 7. finalize in place
    {
        size_t total = BT * S_DIM;
        finalize_kernel<<<(unsigned)(total / 256), 256, 0, stream>>>(
            out, stats, gates, mask_src, steps);
    }
}
